// GINConvNet_83425444758050
// MI455X (gfx1250) — matrix operand order repaired and checked by reading, not compiled, not run on hardware
//
#include <hip/hip_runtime.h>
#include <hip/hip_bf16.h>
#include <stdint.h>

// ---------------- problem constants ----------------
#define NN 65536
#define EE 524288
#define BB 80
#define DD 128
#define BN_EPS 1e-5f

#define LDS_STRIDE 136   // padded bf16 row stride (272B = 16B-aligned, bank-conflict-free)

typedef __bf16 bf16_t;
typedef __attribute__((ext_vector_type(16))) __bf16 v16bf;
typedef __attribute__((ext_vector_type(8)))  __bf16 bf16x8;
typedef __attribute__((ext_vector_type(4)))  __bf16 bf16x4;
typedef __attribute__((ext_vector_type(8)))  float  v8f;

union AFrag { v16bf v; bf16x8 h[2]; __bf16 e[16]; };

static __device__ __forceinline__ void atomic_add_f32(float* p, float v) {
    __hip_atomic_fetch_add(p, v, __ATOMIC_RELAXED, __HIP_MEMORY_SCOPE_AGENT);
}

// ---------------- small elementwise kernels ----------------
__global__ void k_f2bf_t(const float* __restrict__ src, bf16_t* __restrict__ dst, int n) {
    // per-128x128-matrix transpose: dst[mat][n][k] = src[mat][k][n]
    int i = blockIdx.x * blockDim.x + threadIdx.x;
    if (i >= n) return;
    int mat = i >> 14;            // /16384
    int r   = i & 16383;
    int col = r >> 7;             // n (output row)
    int k   = r & 127;            // k (output col)
    dst[i] = (bf16_t)src[(size_t)mat * 16384 + (size_t)k * DD + col];
}

__global__ void k_init_h(const float* __restrict__ x, float* __restrict__ h,
                         bf16_t* __restrict__ h_bf) {
    int i = blockIdx.x * blockDim.x + threadIdx.x;
    float v = x[i];
    h[i] = v;
    h_bf[i] = (bf16_t)v;
}

// gather neighbor features (bf16) and scatter-add into agg (f32 atomics).
// one thread handles 4 features of one edge -> 8-byte bf16x4 gather.
__global__ void k_scatter(const bf16_t* __restrict__ h_bf,
                          const int* __restrict__ src, const int* __restrict__ dst,
                          float* __restrict__ agg) {
    int tid = blockIdx.x * blockDim.x + threadIdx.x; // E*32 threads
    int e  = tid >> 5;
    int c4 = (tid & 31) << 2;
    int s = src[e];
    int d = dst[e];
    bf16x4 v = *(const bf16x4*)(h_bf + (size_t)s * DD + c4);
    float* base = agg + (size_t)d * DD + c4;
#pragma unroll
    for (int j = 0; j < 4; ++j) atomic_add_f32(base + j, (float)v[j]);
}

__global__ void k_make_z(const float* __restrict__ h, const float* __restrict__ agg,
                         bf16_t* __restrict__ z_bf) {
    int i = blockIdx.x * blockDim.x + threadIdx.x;
    z_bf[i] = (bf16_t)(h[i] + agg[i]);
}

__global__ void k_comb_add(float* __restrict__ comb, const float* __restrict__ h) {
    int i = blockIdx.x * blockDim.x + threadIdx.x;
    comb[i] += h[i];
}

// ---------------- WMMA GEMM: [M,128] x [128,128], M multiple of 64 ----------------
// Wt is the 128x128 weight matrix stored TRANSPOSED ([n][k], bf16).
// MODE 0: out(bf16) = relu(A@W + bias)
// MODE 1: out(f32)  = relu(A@W + bias)
// MODE 2: out(f32) += A@W        (no bias / relu)
template <int MODE>
__global__ void k_gemm128(const bf16_t* __restrict__ A, const bf16_t* __restrict__ Wt,
                          const float* __restrict__ bias, void* __restrict__ out) {
    __shared__ bf16_t Wl[DD * LDS_STRIDE];   // ~34 KB, padded rows
    __shared__ float  biasl[DD];

    const int t = threadIdx.x;       // 128 threads = 4 waves (wave32)
    // stage transposed weights into padded LDS rows, coalesced dwords
    {
        const uint32_t* wsrc = (const uint32_t*)Wt;
        uint32_t* wdst = (uint32_t*)Wl;
#pragma unroll
        for (int i = t; i < DD * DD / 2; i += 128) {
            int row = i >> 6;        // n
            int col = i & 63;        // dword within row
            wdst[row * (LDS_STRIDE / 2) + col] = wsrc[i];
        }
    }
    if constexpr (MODE != 2) { if (t < DD) biasl[t] = bias[t]; }
    __syncthreads();

    const int wave = t >> 5;
    const int lane = t & 31;
    const int g = lane >> 4;         // lane group (0/1)
    const int m = lane & 15;         // A row within tile / C col
    const size_t row = (size_t)blockIdx.x * 64 + wave * 16 + m;

    // preload 4 K-chunks of A fragment (two contiguous 8-halfword runs per lane)
    AFrag a[4];
#pragma unroll
    for (int kk = 0; kk < 4; ++kk) {
        const bf16_t* ap = A + row * DD + kk * 32;
        a[kk].h[0] = *(const bf16x8*)(ap + 8 * g);
        a[kk].h[1] = *(const bf16x8*)(ap + 16 + 8 * g);
    }

    const size_t crow0 = (size_t)blockIdx.x * 64 + wave * 16 + 8 * g;

#pragma unroll
    for (int ct = 0; ct < 8; ++ct) {            // 8 column tiles of 16
        v8f acc = {};
#pragma unroll
        for (int kk = 0; kk < 4; ++kk) {
            // B fragment: lane holds W[k = kk*32+8g .. +7 and kk*32+16+8g .. +7][n = ct*16+m],
            // two 8-element runs in transposed storage -> two 16B LDS loads
            AFrag b;
            const bf16_t* bp = Wl + (size_t)(ct * 16 + m) * LDS_STRIDE + kk * 32 + 8 * g;
            b.h[0] = *(const bf16x8*)bp;
            b.h[1] = *(const bf16x8*)(bp + 16);
            acc = __builtin_amdgcn_wmma_f32_16x16x32_bf16(
                false, a[kk].v, false, b.v, (short)0, acc, false, false);
        }
        const size_t obase = crow0 * DD + ct * 16 + m;
        if constexpr (MODE == 0) {
            bf16_t* o = (bf16_t*)out;
            float bb = biasl[ct * 16 + m];
#pragma unroll
            for (int r = 0; r < 8; ++r)
                o[obase + (size_t)r * DD] = (bf16_t)fmaxf(acc[r] + bb, 0.0f);
        } else if constexpr (MODE == 1) {
            float* o = (float*)out;
            float bb = biasl[ct * 16 + m];
#pragma unroll
            for (int r = 0; r < 8; ++r)
                o[obase + (size_t)r * DD] = fmaxf(acc[r] + bb, 0.0f);
        } else {
            float* o = (float*)out;
#pragma unroll
            for (int r = 0; r < 8; ++r)
                o[obase + (size_t)r * DD] += acc[r];
        }
    }
}

// ---------------- batch-norm statistics ----------------
__global__ void k_col_stats(const float* __restrict__ u, float* __restrict__ sums,
                            int rows_per_block) {
    int c = threadIdx.x;                       // 128 threads
    int r0 = blockIdx.x * rows_per_block;
    float s = 0.f, q = 0.f;
    for (int r = r0; r < r0 + rows_per_block; ++r) {
        float v = u[(size_t)r * DD + c];
        s += v;
        q += v * v;
    }
    atomic_add_f32(&sums[c], s);
    atomic_add_f32(&sums[DD + c], q);
}

__global__ void k_bn_finalize(const float* __restrict__ sums,
                              const float* __restrict__ g, const float* __restrict__ b,
                              float* __restrict__ scale, float* __restrict__ shift) {
    int c = threadIdx.x;
    float mean = sums[c] * (1.0f / NN);
    float var = sums[DD + c] * (1.0f / NN) - mean * mean;
    float sc = g[c] * rsqrtf(var + BN_EPS);
    scale[c] = sc;
    shift[c] = b[c] - mean * sc;
}

__global__ void k_bn_apply(const float* __restrict__ u, const float* __restrict__ scale,
                           const float* __restrict__ shift, float* __restrict__ h,
                           bf16_t* __restrict__ h_bf) {
    int i = blockIdx.x * blockDim.x + threadIdx.x;
    int c = i & (DD - 1);
    float v = fmaf(u[i], scale[c], shift[c]);
    h[i] = v;
    h_bf[i] = (bf16_t)v;
}

// ---------------- pooling + head ----------------
__global__ void k_pool(const float* __restrict__ comb, const int* __restrict__ batch,
                       float* __restrict__ pooled) {
    int i = blockIdx.x * blockDim.x + threadIdx.x; // N*128
    int n = i >> 7;
    int c = i & (DD - 1);
    atomic_add_f32(&pooled[(size_t)batch[n] * DD + c], comb[i]);
}

__global__ void k_head_gemm(const float* __restrict__ A, const float* __restrict__ W,
                            const float* __restrict__ bias, float* __restrict__ out,
                            int M, int K, int Nc, int relu) {
    int idx = blockIdx.x * blockDim.x + threadIdx.x;
    if (idx >= M * Nc) return;
    int m = idx / Nc, n = idx % Nc;
    float acc = bias[n];
    for (int k = 0; k < K; ++k) acc += A[(size_t)m * K + k] * W[(size_t)k * Nc + n];
    out[idx] = relu ? fmaxf(acc, 0.0f) : acc;
}

__global__ void k_copy(const float* __restrict__ s, float* __restrict__ d, int n) {
    int i = blockIdx.x * blockDim.x + threadIdx.x;
    if (i < n) d[i] = s[i];
}

// ---------------- launcher ----------------
extern "C" void kernel_launch(void* const* d_in, const int* in_sizes, int n_in,
                              void* d_out, int out_size, void* d_ws, size_t ws_size,
                              hipStream_t stream) {
    (void)in_sizes; (void)n_in; (void)out_size; (void)ws_size;
    const float* x      = (const float*)d_in[0];
    const int*   eidx   = (const int*)d_in[1];
    const int*   batch  = (const int*)d_in[2];
    const float* convW1 = (const float*)d_in[3];
    const float* convb1 = (const float*)d_in[4];
    const float* convW2 = (const float*)d_in[5];
    const float* convb2 = (const float*)d_in[6];
    const float* bn_g   = (const float*)d_in[7];
    const float* bn_b   = (const float*)d_in[8];
    const float* W123   = (const float*)d_in[9];
    const float* Wxd    = (const float*)d_in[10];
    const float* bxd    = (const float*)d_in[11];
    const float* Wfc1   = (const float*)d_in[12];
    const float* bfc1   = (const float*)d_in[13];
    const float* Wfc2   = (const float*)d_in[14];
    const float* bfc2   = (const float*)d_in[15];
    const float* Wout   = (const float*)d_in[16];
    const float* bout   = (const float*)d_in[17];
    const int* src = eidx;
    const int* dst = eidx + EE;
    float* outp = (float*)d_out;

    // carve workspace
    char* ws = (char*)d_ws;
    auto carve = [&](size_t bytes) {
        char* p = ws;
        ws += (bytes + 255) & ~(size_t)255;
        return p;
    };
    const size_t NF4 = (size_t)NN * DD * 4;
    const size_t NF2 = (size_t)NN * DD * 2;
    float*  h     = (float*)carve(NF4);
    float*  agg   = (float*)carve(NF4);   // reused as u (GEMM2 output)
    float*  comb  = (float*)carve(NF4);
    bf16_t* h_bf  = (bf16_t*)carve(NF2);
    bf16_t* z_bf  = (bf16_t*)carve(NF2);
    bf16_t* t_bf  = (bf16_t*)carve(NF2);
    bf16_t* W1bf  = (bf16_t*)carve(5 * DD * DD * 2);   // transposed
    bf16_t* W2bf  = (bf16_t*)carve(5 * DD * DD * 2);   // transposed
    bf16_t* W3bf  = (bf16_t*)carve(3 * DD * DD * 2);   // transposed
    float*  stats = (float*)carve(2 * DD * 4);
    float*  scale = (float*)carve(DD * 4);
    float*  shift = (float*)carve(DD * 4);
    float*  pooled= (float*)carve((size_t)BB * DD * 4);
    float*  hxd   = (float*)carve((size_t)BB * DD * 4);
    float*  hc1   = (float*)carve((size_t)BB * 1024 * 4);
    float*  hc2   = (float*)carve((size_t)BB * 256 * 4);

    const int TB = 256;
    const int elemBlocks = (NN * DD) / TB;

    // weights -> bf16, transposed per 128x128 matrix
    k_f2bf_t<<<(5 * DD * DD + TB - 1) / TB, TB, 0, stream>>>(convW1, W1bf, 5 * DD * DD);
    k_f2bf_t<<<(5 * DD * DD + TB - 1) / TB, TB, 0, stream>>>(convW2, W2bf, 5 * DD * DD);
    k_f2bf_t<<<(3 * DD * DD + TB - 1) / TB, TB, 0, stream>>>(W123,  W3bf, 3 * DD * DD);

    k_init_h<<<elemBlocks, TB, 0, stream>>>(x, h, h_bf);
    hipMemsetAsync(comb, 0, NF4, stream);

    const int gemmBlocks = NN / 64; // 4 waves * 16 rows per block

    for (int i = 0; i < 5; ++i) {
        hipMemsetAsync(agg, 0, NF4, stream);
        k_scatter<<<(EE * 32) / TB, TB, 0, stream>>>(h_bf, src, dst, agg);
        k_make_z<<<elemBlocks, TB, 0, stream>>>(h, agg, z_bf);

        hipMemsetAsync(stats, 0, 2 * DD * 4, stream);
        k_gemm128<0><<<gemmBlocks, 128, 0, stream>>>(z_bf, W1bf + (size_t)i * DD * DD,
                                                     convb1 + i * DD, (void*)t_bf);
        k_gemm128<1><<<gemmBlocks, 128, 0, stream>>>(t_bf, W2bf + (size_t)i * DD * DD,
                                                     convb2 + i * DD, (void*)agg);
        k_col_stats<<<256, DD, 0, stream>>>(agg, stats, NN / 256);
        k_bn_finalize<<<1, DD, 0, stream>>>(stats, bn_g + i * DD, bn_b + i * DD, scale, shift);
        k_bn_apply<<<elemBlocks, TB, 0, stream>>>(agg, scale, shift, h, h_bf);

        if (i == 0 || i == 4) {
            k_comb_add<<<elemBlocks, TB, 0, stream>>>(comb, h);
        } else {
            k_gemm128<2><<<gemmBlocks, 128, 0, stream>>>(h_bf, W3bf + (size_t)(i - 1) * DD * DD,
                                                         nullptr, (void*)comb);
        }
    }

    // global add pool over batch ids
    hipMemsetAsync(pooled, 0, (size_t)BB * DD * 4, stream);
    k_pool<<<elemBlocks, TB, 0, stream>>>(comb, batch, pooled);

    // head MLP (tiny, fp32)
    k_head_gemm<<<(BB * DD + TB - 1) / TB, TB, 0, stream>>>(pooled, Wxd, bxd, hxd, BB, DD, DD, 1);
    k_head_gemm<<<(BB * 1024 + TB - 1) / TB, TB, 0, stream>>>(hxd, Wfc1, bfc1, hc1, BB, DD, 1024, 1);
    k_head_gemm<<<(BB * 256 + TB - 1) / TB, TB, 0, stream>>>(hc1, Wfc2, bfc2, hc2, BB, 1024, 256, 1);
    k_head_gemm<<<(BB + TB - 1) / TB, TB, 0, stream>>>(hc2, Wout, bout, outp, BB, 256, 1, 0);

    // x_protein = pooled reshaped, appended after out
    k_copy<<<(BB * DD + TB - 1) / TB, TB, 0, stream>>>(pooled, outp + BB, BB * DD);
}
